// TFT_ActorCritic_Model_62723702391477
// MI455X (gfx1250) — compile-verified, matrix operand order checked by reading, not run on hardware
//
#include <hip/hip_runtime.h>

// ---------------------------------------------------------------------------
// TFT Actor-Critic forward for MI455X (gfx1250, wave32, WMMA).
// All heavy matmuls via v_wmma_f32_16x16x32_f16 (f32 accumulate).
// WMMA operands stored in fragment-native swizzled LDS layout so every
// fragment load is one contiguous 32B read (2x ds_load_b128, no packing VALU).
// Weight panels for the GEMM/LSTM kernels are staged into LDS with the
// Tensor Data Mover (tensor_load_to_lds + s_wait_tensorcnt), 6-arg builtin
// (amdgpu-toolchain / clang-23 arity, confirmed by compile probe).
// ---------------------------------------------------------------------------

#define BB 2048
#define SS 16
#define FF 32
#define HH 64
#define AA 3
#define RR (BB*SS)   // 32768 (b,s) rows

typedef __attribute__((ext_vector_type(16))) _Float16 v16h;
typedef __attribute__((ext_vector_type(8)))  float    v8f;
typedef unsigned int u32x4 __attribute__((ext_vector_type(4)));
typedef int          i32x8 __attribute__((ext_vector_type(8)));
typedef int          i32x4 __attribute__((ext_vector_type(4)));

#define WMMA(a,b,c) __builtin_amdgcn_wmma_f32_16x16x32_f16(false,(a),false,(b),(short)0,(c),false,false)

#if defined(__has_builtin)
#if __has_builtin(__builtin_amdgcn_tensor_load_to_lds) && __has_builtin(__builtin_amdgcn_s_wait_tensorcnt)
#define USE_TDM 1
#endif
#endif

__device__ __forceinline__ float sigmoidf_(float x){ return 1.f/(1.f+__expf(-x)); }
__device__ __forceinline__ float eluf_(float x){ return x>0.f ? x : (__expf(x)-1.f); }

// Fragment-native swizzled element index.
// ISA 16-bit operand layout (A 16x32 / B 32x16 are symmetric):
//   lane = ((kl&15)>>3)*16 + rowlane   (rowlane = m&15 for A, n&15 for B)
//   within-lane element i = (kl&7) + ((kl>>4)<<3)     (kl = k within 32-tile)
// Stored so each (tile,lane) owns 16 contiguous halfs -> one 32B load.
__device__ __forceinline__ int swzi(int mt, int kt, int ml, int kl, int KT){
  int lane = (((kl & 15) >> 3) << 4) + ml;
  return ((((mt*KT + kt) << 5) + lane) << 4) + (kl & 7) + ((kl >> 4) << 3);
}

__device__ __forceinline__ v16h frag_ld(const _Float16* __restrict__ base,
                                        int tile, int lane){
  return *(const v16h*)(base + ((((tile << 5) + lane)) << 4));
}

// TDM: DMA one flat f16 weight panel (nelem elements) from global to LDS.
// D# built per CDNA5 ISA ch.8: group0 = {count=1 | lds_addr | global_addr |
// type=2}; group1 = {mask=0, data_size=1(2B), tensor_dim0=nelem, tensor_dim1=1,
// tile_dim0=nelem, tile_dim1=1, dim0_stride=nelem}; groups 2/3 unused (2D).
// LDS byte offset = low 32 bits of the flat shared-pointer address (aperture).
__device__ __forceinline__ void tdm_load_panel(const _Float16* __restrict__ gsrc,
                                               _Float16* lds, unsigned nelem){
#ifdef USE_TDM
  unsigned long long ga = (unsigned long long)(size_t)gsrc;
  unsigned lo = (unsigned)(size_t)(void*)lds;      // LDS byte offset
  u32x4 g0;
  g0[0] = 1u;                                      // count=1, user mode
  g0[1] = lo;                                      // lds_addr
  g0[2] = (unsigned)ga;                            // global_addr[31:0]
  g0[3] = ((unsigned)(ga >> 32) & 0x01FFFFFFu) | 0x80000000u; // addr[56:32]|type=2
  i32x8 g1;
  g1[0] = (int)(1u << 16);                         // data_size=1 (2 bytes)
  g1[1] = (int)(nelem << 16);                      // tensor_dim0[15:0] @ bits63:48
  g1[2] = (int)((nelem >> 16) | (1u << 16));       // dim0 hi | tensor_dim1=1
  g1[3] = (int)(nelem << 16);                      // tile_dim0 @ bits127:112
  g1[4] = 1;                                       // tile_dim1=1, tile_dim2=0
  g1[5] = (int)nelem;                              // tensor_dim0_stride lo
  g1[6] = 0;
  g1[7] = 0;
  i32x4 gz4 = {0,0,0,0};
  i32x8 gz8 = {0,0,0,0,0,0,0,0};
  __builtin_amdgcn_tensor_load_to_lds(g0, g1, gz4, gz4, gz8, 0);
  __builtin_amdgcn_s_wait_tensorcnt(0);            // issuing wave waits; barrier follows
#else
  (void)gsrc; (void)lds; (void)nelem;
#endif
}

// ---------------------------------------------------------------------------
// Weight prep: f32 src [N][64] (row-major, out_features x 64) ->
// swizzled f16 B-operand (K=64 => 2 k-tiles per n-tile).
// ---------------------------------------------------------------------------
__global__ void k_w2h_sw(const float* __restrict__ src, _Float16* __restrict__ dst,
                         int N){
  int j = blockIdx.x*256 + threadIdx.x;
  if (j >= 64*N) return;
  int i = j & 15, lane = (j >> 4) & 31, t = j >> 9;
  int kt = t & 1, nt = t >> 1;
  int k = kt*32 + ((lane >> 4) << 3) + (i & 7) + ((i >> 3) << 4);
  int n = nt*16 + (lane & 15);
  dst[j] = (_Float16)src[n*64 + k];
}

// ---------------------------------------------------------------------------
// wg GRN + softmax over F=32 -> variable-selection weights. One wave per row,
// F==32 lanes -> pure shfl VALU kernel (0.2 GF total, not worth WMMA).
// ---------------------------------------------------------------------------
__global__ void __launch_bounds__(256) k_wg(
    const float* __restrict__ state,
    const float* __restrict__ w1, const float* __restrict__ b1,
    const float* __restrict__ w2, const float* __restrict__ b2,
    const float* __restrict__ g,  const float* __restrict__ bt,
    float* __restrict__ wout){
  int row  = (int)((blockIdx.x*blockDim.x + threadIdx.x) >> 5);
  int lane = threadIdx.x & 31;
  if (row >= RR) return;
  float s = state[(size_t)row*FF + lane];
  float acc = b1[lane];
  #pragma unroll
  for (int f=0; f<FF; ++f) acc += __shfl(s, f, 32) * w1[lane*FF + f];
  float h1 = eluf_(acc);
  float a0 = b2[lane], a1 = b2[lane+32];
  #pragma unroll
  for (int f=0; f<FF; ++f){
    float hf = __shfl(h1, f, 32);
    a0 += hf*w2[lane*FF+f]; a1 += hf*w2[(lane+32)*FF+f];
  }
  float y = s + a0 * sigmoidf_(a1);          // wg has no skip proj: res = x
  float m = y;
  for (int o=16;o>0;o>>=1) m += __shfl_xor(m,o,32);
  m *= (1.f/FF);
  float d = y - m, v = d*d;
  for (int o=16;o>0;o>>=1) v += __shfl_xor(v,o,32);
  v *= (1.f/FF);
  float ln = d * rsqrtf(v + 1e-5f) * g[lane] + bt[lane];
  float mx = ln;
  for (int o=16;o>0;o>>=1) mx = fmaxf(mx, __shfl_xor(mx,o,32));
  float e = __expf(ln - mx), sum = e;
  for (int o=16;o>0;o>>=1) sum += __shfl_xor(sum,o,32);
  wout[(size_t)row*FF + lane] = e / sum;
}

// ---------------------------------------------------------------------------
// fg GRN (17 GFLOP GEMM, 85% of the model) + softmax-weighted einsum -> x.
// 256 thr = 8 waves, 4 (b,s) groups = 128 f-rows. 16 WMMA / wave. GLU+LN in
// accumulators (h and h+64 sit in tiles nt / nt+4 of the same lane).
// ---------------------------------------------------------------------------
__global__ void __launch_bounds__(256) k_fg(
    const float* __restrict__ state, const float* __restrict__ wsm,
    const float* __restrict__ fc1w, const float* __restrict__ fc1b,
    const _Float16* __restrict__ w2t, const float* __restrict__ fc2b,
    const float* __restrict__ skw, const float* __restrict__ skb,
    const float* __restrict__ lng, const float* __restrict__ lnb,
    float* __restrict__ xout){
  __shared__ __align__(32) _Float16 lA[128*64];   // swizzled A (8 mt x 2 kt)
  __shared__ __align__(32) _Float16 lW[64*128];   // swizzled B (8 nt x 2 kt)
  __shared__ float lV[128];
  __shared__ float lWgt[128];
  __shared__ float lXp[8*64];
  int tid  = threadIdx.x;
  int row0 = blockIdx.x*4;                    // (b,s) base
#ifdef USE_TDM
  if (tid < 32) tdm_load_panel(w2t, lW, 64*128);
#else
  for (int i=tid;i<(64*128)>>3;i+=256) ((uint4*)lW)[i] = ((const uint4*)w2t)[i];
#endif
  for (int i=tid;i<128;i+=256){
    lV[i]   = state[(size_t)row0*FF + i];
    lWgt[i] = wsm  [(size_t)row0*FF + i];
  }
  __syncthreads();
  for (int i=tid;i<128*64;i+=256){
    int m=i>>6, k=i&63;
    lA[swzi(m>>4, k>>5, m&15, k&31, 2)] =
        (_Float16)eluf_(lV[m]*fc1w[k] + fc1b[k]);
  }
  __syncthreads();
  int wv=tid>>5, lane=tid&31, half=lane>>4, nl=lane&15;
  int m0 = wv<<4;
  v16h a0 = frag_ld(lA, (wv<<1)+0, lane);
  v16h a1 = frag_ld(lA, (wv<<1)+1, lane);
  v8f acc[8];
  #pragma unroll
  for (int nt=0;nt<8;++nt){
    v8f c = {};
    c = WMMA(a0, frag_ld(lW, (nt<<1)+0, lane), c);
    c = WMMA(a1, frag_ld(lW, (nt<<1)+1, lane), c);
    acc[nt] = c;
  }
  float pacc[4] = {0.f,0.f,0.f,0.f};
  #pragma unroll
  for (int r=0;r<8;++r){
    int m = m0 + r + 8*half;                  // local f-row 0..127
    float v = lV[m], wt = lWgt[m];
    float y[4]; float sum = 0.f;
    #pragma unroll
    for (int nt=0;nt<4;++nt){
      int hh = nt*16 + nl;
      float x1 = acc[nt][r]   + fc2b[hh];
      float x2 = acc[nt+4][r] + fc2b[64+hh];
      float yy = v*skw[hh] + skb[hh] + x1*sigmoidf_(x2);
      y[nt] = yy; sum += yy;
    }
    for (int o=8;o>0;o>>=1) sum += __shfl_xor(sum,o,32);   // intra-half (row)
    float mean = sum*(1.f/64.f);
    float d[4]; float var = 0.f;
    #pragma unroll
    for (int nt=0;nt<4;++nt){ d[nt]=y[nt]-mean; var += d[nt]*d[nt]; }
    for (int o=8;o>0;o>>=1) var += __shfl_xor(var,o,32);
    float rstd = rsqrtf(var*(1.f/64.f) + 1e-5f);
    #pragma unroll
    for (int nt=0;nt<4;++nt){
      int hh = nt*16+nl;
      pacc[nt] += wt*(d[nt]*rstd*lng[hh] + lnb[hh]);
    }
  }
  #pragma unroll
  for (int nt=0;nt<4;++nt){
    float tot = pacc[nt] + __shfl_xor(pacc[nt],16,32);     // join the 2 halves
    if (half==0) lXp[wv*64 + nt*16 + nl] = tot;
  }
  __syncthreads();
  { int g=tid>>6, hh=tid&63;                 // 4 groups * 64 = 256 threads
    xout[((size_t)row0+g)*64 + hh] = lXp[(2*g)*64+hh] + lXp[(2*g+1)*64+hh]; }
}

// ---------------------------------------------------------------------------
// Generic (rows x 64) @ (64 x N) WMMA GEMM with bias. Used for the hoisted
// x@wih.T precompute (N=256, bias=bih+bhh) and qkv (N=192, bias=attn_in_b).
// ---------------------------------------------------------------------------
__global__ void __launch_bounds__(256) k_gemm(
    const float* __restrict__ src, const _Float16* __restrict__ Wt,
    const float* __restrict__ bias0, const float* __restrict__ bias1,
    float* __restrict__ dst, int N){
  __shared__ __align__(32) _Float16 lA[128*64];
  __shared__ __align__(32) _Float16 lW[64*256];
  int tid = threadIdx.x;
  size_t row0 = (size_t)blockIdx.x*128;
#ifdef USE_TDM
  if (tid < 32) tdm_load_panel(Wt, lW, (unsigned)(64*N));
#else
  for (int i=tid;i<(64*N)>>3;i+=256) ((uint4*)lW)[i] = ((const uint4*)Wt)[i];
#endif
  for (int i=tid;i<128*64;i+=256){
    int m=i>>6, k=i&63;
    lA[swzi(m>>4, k>>5, m&15, k&31, 2)] = (_Float16)src[row0*64 + i];
  }
  if ((int)blockIdx.x + 1 < (int)gridDim.x)
    __builtin_prefetch(src + (row0+128)*64, 0, 1);   // global_prefetch_b8
  __syncthreads();
  int wv=tid>>5, lane=tid&31, half=lane>>4, nl=lane&15;
  int m0 = wv<<4;
  v16h a0 = frag_ld(lA, (wv<<1)+0, lane);
  v16h a1 = frag_ld(lA, (wv<<1)+1, lane);
  int NT = N >> 4;
  for (int nt=0;nt<NT;++nt){
    v8f c = {};
    c = WMMA(a0, frag_ld(lW, (nt<<1)+0, lane), c);
    c = WMMA(a1, frag_ld(lW, (nt<<1)+1, lane), c);
    int col = nt*16 + nl;
    float bb = bias0[col] + (bias1 ? bias1[col] : 0.f);
    #pragma unroll
    for (int r=0;r<8;++r)
      dst[(row0 + m0 + r + 8*half)*N + col] = c[r] + bb;
  }
}

// ---------------------------------------------------------------------------
// LSTM scan. 128 batch rows per block; h as swizzled f16 in LDS, c in 32
// regs/thread. Gate quadruple i/f/g/o for hidden h lives in tiles
// {nt,nt+4,nt+8,nt+12} of one lane -> update never leaves registers.
// ---------------------------------------------------------------------------
__global__ void __launch_bounds__(256) k_lstm(
    const float* __restrict__ xpre, const _Float16* __restrict__ whhT,
    float* __restrict__ hs){
  __shared__ __align__(32) _Float16 lW[64*256];
  __shared__ __align__(32) _Float16 lH[128*64];
  int tid = threadIdx.x;
  int b0  = blockIdx.x*128;
#ifdef USE_TDM
  if (tid < 32) tdm_load_panel(whhT, lW, 64*256);
#else
  for (int i=tid;i<(64*256)>>3;i+=256) ((uint4*)lW)[i] = ((const uint4*)whhT)[i];
#endif
  for (int i=tid;i<128*64;i+=256) lH[i] = (_Float16)0.f;
  __syncthreads();
  int wv=tid>>5, lane=tid&31, half=lane>>4, nl=lane&15;
  int m0 = wv<<4;
  float c[32];
  #pragma unroll
  for (int i=0;i<32;++i) c[i]=0.f;
  for (int t=0;t<SS;++t){
    v16h a0 = frag_ld(lH, (wv<<1)+0, lane);
    v16h a1 = frag_ld(lH, (wv<<1)+1, lane);
    v8f acc[16];
    #pragma unroll
    for (int nt=0;nt<16;++nt){
      v8f cc = {};
      cc = WMMA(a0, frag_ld(lW, (nt<<1)+0, lane), cc);
      cc = WMMA(a1, frag_ld(lW, (nt<<1)+1, lane), cc);
      acc[nt] = cc;
    }
    __syncthreads();                         // all reads of lH done
    #pragma unroll
    for (int ht=0;ht<4;++ht){
      #pragma unroll
      for (int r=0;r<8;++r){
        int rowl = m0 + r + 8*half;
        size_t brow = (size_t)b0 + rowl;
        int hidx = ht*16 + nl;
        const float* xp = xpre + (brow*SS + (size_t)t)*256;
        float gi = acc[ht][r]    + xp[hidx];
        float gf = acc[ht+4][r]  + xp[hidx+64];
        float gg = acc[ht+8][r]  + xp[hidx+128];
        float go = acc[ht+12][r] + xp[hidx+192];
        int ci = ht*8 + r;
        float cn = sigmoidf_(gf)*c[ci] + sigmoidf_(gi)*tanhf(gg);
        c[ci] = cn;
        float hn = sigmoidf_(go)*tanhf(cn);
        lH[swzi(rowl>>4, hidx>>5, rowl&15, hidx&31, 2)] = (_Float16)hn;
        hs[(brow*SS + (size_t)t)*64 + hidx] = hn;
      }
    }
    __syncthreads();
  }
}

// ---------------------------------------------------------------------------
// Attention: scores (1 WMMA / head), register softmax, att@v (2 WMMA), out
// projection (8 WMMA over 4 waves), residual + ln2. Block = 4 batches.
// q/k/v staged directly into swizzled operand layouts; lOut aliases lQ/lKT.
// ---------------------------------------------------------------------------
__global__ void __launch_bounds__(256) k_attn(
    const float* __restrict__ qkv, const float* __restrict__ hs,
    const _Float16* __restrict__ woT, const float* __restrict__ ob,
    const float* __restrict__ g2, const float* __restrict__ b2,
    float* __restrict__ xattn){
  __shared__ __align__(32) char smem[57344];
  _Float16* lQ   = (_Float16*)(smem);            // [8 bh] A 16x32 swz     8KB
  _Float16* lKT  = (_Float16*)(smem + 8192);     // [8 bh] B 32x16 swz     8KB
  _Float16* lVv  = (_Float16*)(smem + 16384);    // [8 bh] B 32(pad)x32 swz 16KB
  _Float16* lAtt = (_Float16*)(smem + 32768);    // [8 bh] A 16x32(pad) swz 8KB
  _Float16* lAO  = (_Float16*)(smem + 40960);    // A 64x64 swz            8KB
  _Float16* lWo  = (_Float16*)(smem + 49152);    // B 64x64 swz            8KB
  float*    lOut = (float*)(smem);               // [64][64] f32, aliases lQ/lKT
  int tid = threadIdx.x;
  int b0  = blockIdx.x*4;
  for (int i=tid;i<(64*64)>>3;i+=256) ((uint4*)lWo)[i] = ((const uint4*)woT)[i];
  for (int i=tid;i<8*32*32;i+=256) lVv[i]  = (_Float16)0.f;
  for (int i=tid;i<8*16*32;i+=256) lAtt[i] = (_Float16)0.f;
  __syncthreads();
  for (int i=tid;i<4*16*192;i+=256){
    int row = i/192, col = i - row*192;
    float v = qkv[((size_t)b0*16 + row)*192 + col];
    int bl = row>>4, s = row&15;
    if (col < 64){                             // q: A operand, m=s, k=d
      int h=col>>5, d=col&31;
      lQ[((bl*2+h)<<9) + swzi(0,0,s,d,1)] = (_Float16)v;
    } else if (col < 128){                     // k^T: B operand, n=s, k=d
      int cc=col-64; int h=cc>>5, d=cc&31;
      lKT[((bl*2+h)<<9) + swzi(0,0,s,d,1)] = (_Float16)v;
    } else {                                   // v: B operand, n=d, k=s(pad 32)
      int cc=col-128; int h=cc>>5, d=cc&31;
      lVv[((bl*2+h)<<10) + swzi(d>>4,0,d&15,s,1)] = (_Float16)v;
    }
  }
  __syncthreads();
  int wv=tid>>5, lane=tid&31, half=lane>>4, nl=lane&15;
  // ---- scores + softmax (wave wv == (batch-local, head)) ----
  {
    v8f sc = {};
    sc = WMMA(frag_ld(lQ + (wv<<9), 0, lane),
              frag_ld(lKT + (wv<<9), 0, lane), sc);
    const float scale = 0.17677669529663687f;      // 1/sqrt(32)
    #pragma unroll
    for (int r=0;r<8;++r){
      float v = sc[r]*scale;
      float mx = v;
      for (int o=8;o>0;o>>=1) mx = fmaxf(mx, __shfl_xor(mx,o,32));
      float e = __expf(v - mx), sm = e;
      for (int o=8;o>0;o>>=1) sm += __shfl_xor(sm,o,32);
      lAtt[(wv<<9) + swzi(0,0, r+8*half, nl, 1)] = (_Float16)(e/sm);
    }
  }
  __syncthreads();
  // ---- att @ v ----
  {
    v16h aa = frag_ld(lAtt + (wv<<9), 0, lane);
    int bl = wv>>1, hh = wv&1;
    #pragma unroll
    for (int nt=0;nt<2;++nt){
      v8f ao = {};
      ao = WMMA(aa, frag_ld(lVv + (wv<<10), nt, lane), ao);
      #pragma unroll
      for (int r=0;r<8;++r){
        int rown = (bl<<4) + r + 8*half;
        int coln = (hh<<5) + nt*16 + nl;
        lAO[swzi(rown>>4, coln>>5, rown&15, coln&31, 2)] = (_Float16)ao[r];
      }
    }
  }
  __syncthreads();
  // ---- out projection (64 rows, waves 0..3) ----
  if (wv < 4){
    int m0 = wv<<4;
    v16h a0 = frag_ld(lAO, (wv<<1)+0, lane);
    v16h a1 = frag_ld(lAO, (wv<<1)+1, lane);
    #pragma unroll
    for (int nt=0;nt<4;++nt){
      v8f c = {};
      c = WMMA(a0, frag_ld(lWo, (nt<<1)+0, lane), c);
      c = WMMA(a1, frag_ld(lWo, (nt<<1)+1, lane), c);
      int col = nt*16 + nl;
      #pragma unroll
      for (int r=0;r<8;++r) lOut[(m0 + r + 8*half)*64 + col] = c[r] + ob[col];
    }
  }
  __syncthreads();
  // ---- residual + ln2 ----
  #pragma unroll
  for (int rr=0;rr<8;++rr){
    int row = wv*8 + rr;
    size_t grow = (size_t)b0*16 + row;
    float xa = hs[grow*64 + lane]      + lOut[row*64 + lane];
    float xb = hs[grow*64 + 32 + lane] + lOut[row*64 + 32 + lane];
    float sum = xa + xb;
    for (int o=16;o>0;o>>=1) sum += __shfl_xor(sum,o,32);
    float mean = sum*(1.f/64.f);
    float da = xa-mean, db = xb-mean;
    float var = da*da + db*db;
    for (int o=16;o>0;o>>=1) var += __shfl_xor(var,o,32);
    float rstd = rsqrtf(var*(1.f/64.f) + 1e-5f);
    xattn[grow*64 + lane]      = da*rstd*g2[lane]    + b2[lane];
    xattn[grow*64 + 32 + lane] = db*rstd*g2[lane+32] + b2[lane+32];
  }
}

// ---------------------------------------------------------------------------
// Output heads + critics: one wave per batch element, shfl-broadcast matvecs.
// ---------------------------------------------------------------------------
__global__ void __launch_bounds__(256) k_heads(
    const float* __restrict__ xattn, const float* __restrict__ onehot,
    const float* __restrict__ a1w, const float* __restrict__ a1b,
    const float* __restrict__ a2w, const float* __restrict__ a2b,
    const float* __restrict__ a3w, const float* __restrict__ a3b,
    const float* __restrict__ fcw, const float* __restrict__ fcb,
    const float* __restrict__ s1w, const float* __restrict__ s1b,
    const float* __restrict__ s2w, const float* __restrict__ s2b,
    const float* __restrict__ c11w, const float* __restrict__ c11b,
    const float* __restrict__ c12w, const float* __restrict__ c12b,
    const float* __restrict__ c13w, const float* __restrict__ c13b,
    const float* __restrict__ c21w, const float* __restrict__ c21b,
    const float* __restrict__ c22w, const float* __restrict__ c22b,
    const float* __restrict__ c23w, const float* __restrict__ c23b,
    float* __restrict__ out){
  int b    = (int)((blockIdx.x*blockDim.x + threadIdx.x) >> 5);
  int lane = threadIdx.x & 31;
  if (b >= BB) return;
  size_t crow = ((size_t)b*SS + (SS-1))*64;
  float c0 = xattn[crow + lane], c1 = xattn[crow + 32 + lane];
  auto ctxk = [&](int k)->float{ return __shfl(k<32 ? c0 : c1, k&31, 32); };

  // policy: a1 (H->2H) relu, a2 (2H->H) relu, a3 (H->A)
  float h0=a1b[lane], h1=a1b[lane+32], h2=a1b[lane+64], h3=a1b[lane+96];
  for (int k=0;k<64;++k){ float ck=ctxk(k);
    h0+=ck*a1w[lane*64+k];      h1+=ck*a1w[(lane+32)*64+k];
    h2+=ck*a1w[(lane+64)*64+k]; h3+=ck*a1w[(lane+96)*64+k]; }
  h0=fmaxf(h0,0.f); h1=fmaxf(h1,0.f); h2=fmaxf(h2,0.f); h3=fmaxf(h3,0.f);
  float g0=a2b[lane], g1=a2b[lane+32];
  for (int k=0;k<128;++k){
    float hv = __shfl(k<32?h0 : k<64?h1 : k<96?h2 : h3, k&31, 32);
    g0+=hv*a2w[lane*128+k]; g1+=hv*a2w[(lane+32)*128+k];
  }
  g0=fmaxf(g0,0.f); g1=fmaxf(g1,0.f);
  float pol = (lane<AA) ? a3b[lane] : 0.f;
  for (int k=0;k<64;++k){
    float gv = __shfl(k<32?g0:g1, k&31, 32);
    if (lane<AA) pol += gv*a3w[lane*64+k];
  }
  if (lane<AA) out[(size_t)b*AA + lane] = pol;

  // forecast (H->F), F==32 lanes
  float fo = fcb[lane];
  for (int k=0;k<64;++k) fo += ctxk(k)*fcw[lane*64+k];
  out[10240 + (size_t)b*FF + lane] = fo;

  // safety: s1 relu, s2 sigmoid
  float t0=s1b[lane], t1=s1b[lane+32];
  for (int k=0;k<64;++k){ float ck=ctxk(k); t0+=ck*s1w[lane*64+k]; t1+=ck*s1w[(lane+32)*64+k]; }
  t0=fmaxf(t0,0.f); t1=fmaxf(t1,0.f);
  float sa = (lane<AA) ? s2b[lane] : 0.f;
  for (int k=0;k<64;++k){
    float tv = __shfl(k<32?t0:t1, k&31, 32);
    if (lane<AA) sa += tv*s2w[lane*64+k];
  }
  if (lane<AA) out[75776 + (size_t)b*AA + lane] = sigmoidf_(sa);

  // critics over ca = [ctx(64) | onehot(3)]
  auto cak = [&](int k)->float{ return k<64 ? ctxk(k) : onehot[(size_t)b*AA + (k-64)]; };
  const float* W1[2]={c11w,c21w}; const float* B1[2]={c11b,c21b};
  const float* W2[2]={c12w,c22w}; const float* B2[2]={c12b,c22b};
  const float* W3[2]={c13w,c23w}; const float* B3[2]={c13b,c23b};
  float vals[2];
  for (int ci=0;ci<2;++ci){
    const float* w1=W1[ci]; const float* b1=B1[ci];
    const float* w2=W2[ci]; const float* b2=B2[ci];
    float q0=b1[lane], q1=b1[lane+32], q2=b1[lane+64], q3=b1[lane+96];
    for (int k=0;k<67;++k){ float v=cak(k);
      q0+=v*w1[lane*67+k];      q1+=v*w1[(lane+32)*67+k];
      q2+=v*w1[(lane+64)*67+k]; q3+=v*w1[(lane+96)*67+k]; }
    q0=fmaxf(q0,0.f); q1=fmaxf(q1,0.f); q2=fmaxf(q2,0.f); q3=fmaxf(q3,0.f);
    float r0=b2[lane], r1=b2[lane+32];
    for (int k=0;k<128;++k){
      float v=__shfl(k<32?q0 : k<64?q1 : k<96?q2 : q3, k&31, 32);
      r0+=v*w2[lane*128+k]; r1+=v*w2[(lane+32)*128+k];
    }
    r0=fmaxf(r0,0.f); r1=fmaxf(r1,0.f);
    float part = r0*W3[ci][lane] + r1*W3[ci][lane+32];
    for (int o=16;o>0;o>>=1) part += __shfl_xor(part,o,32);
    vals[ci] = part + B3[ci][0];
  }
  if (lane==0){ out[6144 + b] = vals[0]; out[8192 + b] = vals[1]; }
}

// ---------------------------------------------------------------------------
extern "C" void kernel_launch(void* const* d_in, const int* in_sizes, int n_in,
                              void* d_out, int out_size, void* d_ws, size_t ws_size,
                              hipStream_t stream){
  (void)in_sizes; (void)n_in; (void)out_size; (void)ws_size;
  const float* state  = (const float*)d_in[0];
  const float* onehot = (const float*)d_in[1];
  #define P(i) ((const float*)d_in[(i)])

  char* w = (char*)d_ws;
  auto carve = [&](size_t bytes)->void*{
    void* p = (void*)w; w += (bytes + 255) & ~(size_t)255; return p; };
  float* wsm    = (float*)carve((size_t)RR*FF*4);     // softmax weights
  float* x      = (float*)carve((size_t)RR*64*4);     // LSTM input
  float* xpre   = (float*)carve((size_t)RR*256*4);    // x @ wih.T + biases
  float* hsbuf  = (float*)carve((size_t)RR*64*4);     // LSTM hidden seq
  float* qkvb   = (float*)carve((size_t)RR*192*4);    // qkv
  float* xattn  = (float*)carve((size_t)RR*64*4);     // post-attn LN
  _Float16* fg2T = (_Float16*)carve(64*128*2);
  _Float16* wihT = (_Float16*)carve(64*256*2);
  _Float16* whhT = (_Float16*)carve(64*256*2);
  _Float16* inT  = (_Float16*)carve(64*192*2);
  _Float16* outT = (_Float16*)carve(64*64*2);

  // weight f16 swizzled transposes
  k_w2h_sw<<<(64*128+255)/256,256,0,stream>>>(P(10), fg2T, 128);
  k_w2h_sw<<<(64*256+255)/256,256,0,stream>>>(P(16), wihT, 256);
  k_w2h_sw<<<(64*256+255)/256,256,0,stream>>>(P(17), whhT, 256);
  k_w2h_sw<<<(64*192+255)/256,256,0,stream>>>(P(20), inT,  192);
  k_w2h_sw<<<(64*64 +255)/256,256,0,stream>>>(P(22), outT, 64);

  k_wg  <<<RR/8,  256, 0, stream>>>(state, P(2),P(3),P(4),P(5),P(6),P(7), wsm);
  k_fg  <<<RR/4,  256, 0, stream>>>(state, wsm, P(8),P(9), fg2T, P(11),
                                    P(14),P(15), P(12),P(13), x);
  k_gemm<<<RR/128,256, 0, stream>>>(x, wihT, P(18), P(19), xpre, 256);
  k_lstm<<<BB/128,256, 0, stream>>>(xpre, whhT, hsbuf);
  k_gemm<<<RR/128,256, 0, stream>>>(hsbuf, inT, P(21), nullptr, qkvb, 192);
  k_attn<<<BB/4,  256, 0, stream>>>(qkvb, hsbuf, outT, P(23), P(24),P(25), xattn);
  k_heads<<<BB/8, 256, 0, stream>>>(xattn, onehot,
      P(26),P(27),P(28),P(29),P(30),P(31),
      P(32),P(33),P(34),P(35),P(36),P(37),
      P(38),P(39),P(40),P(41),P(42),P(43),
      P(44),P(45),P(46),P(47),P(48),P(49),
      (float*)d_out);
  #undef P
}
